// MHSARPB_52432960749590
// MI455X (gfx1250) — matrix operand order repaired and checked by reading, not compiled, not run on hardware
//
#include <hip/hip_runtime.h>

#define DEV __device__ __forceinline__

typedef __attribute__((ext_vector_type(16))) __bf16        v16bf;
typedef __attribute__((ext_vector_type(8)))  float          v8f;
typedef __attribute__((ext_vector_type(8)))  unsigned short u16x8;

union Frag { v16bf v; u16x8 u[2]; };

static constexpr int S   = 32;
static constexpr int H   = 12;
static constexpr int HD  = 64;
static constexpr int Bsz = 16;
static constexpr int N   = S * S;     // 1024
static constexpr int C   = H * HD;    // 768
static constexpr int C3  = 3 * C;     // 2304
static constexpr int M   = Bsz * N;   // 16384

DEV unsigned short f2bf(float f) {
    union { float f; unsigned u; } c; c.f = f;
    unsigned r = c.u + 0x7FFFu + ((c.u >> 16) & 1u);   // round-to-nearest-even
    return (unsigned short)(r >> 16);
}

// A-matrix 16x32 bf16 fragment from row-major [rows, ldk] memory.
// lane = row; half 0: K 0-15, half 1: K 16-31.
DEV Frag load_afrag(const unsigned short* __restrict__ base, int ldk, int row, int k0, int half) {
    Frag f;
    const unsigned short* p = base + (size_t)row * ldk + k0 + half * 16;
    f.u[0] = *(const u16x8*)p;
    f.u[1] = *(const u16x8*)(p + 8);
    return f;
}

// B-matrix 32x16 bf16 fragment; memory holds B^T row-major: [col][kdim] contiguous in kdim.
// lane = col; half 0: K rows 0-15, half 1: K rows 16-31.
DEV Frag load_bfrag(const unsigned short* __restrict__ base, int ldk, int col, int k0, int half) {
    Frag f;
    const unsigned short* p = base + (size_t)col * ldk + k0 + half * 16;
    f.u[0] = *(const u16x8*)p;
    f.u[1] = *(const u16x8*)(p + 8);
    return f;
}

DEV v8f wmma_bf16(const Frag& a, const Frag& b, v8f c) {
    return __builtin_amdgcn_wmma_f32_16x16x32_bf16(false, a.v, false, b.v, (short)0, c, false, false);
}

// ---------------------------------------------------------------- convert ---
__global__ void convert_bf16(const float* __restrict__ x, const float* __restrict__ wq,
                             const float* __restrict__ wp, unsigned short* __restrict__ xb,
                             unsigned short* __restrict__ wqb, unsigned short* __restrict__ wpb) {
    const int NX = M * C, NQ = C3 * C, NP = C * C;
    for (int i = blockIdx.x * blockDim.x + threadIdx.x; i < NX + NQ + NP;
         i += gridDim.x * blockDim.x) {
        if (i < NX)            xb[i]            = f2bf(x[i]);
        else if (i < NX + NQ)  wqb[i - NX]      = f2bf(wq[i - NX]);
        else                   wpb[i - NX - NQ] = f2bf(wp[i - NX - NQ]);
    }
}

// ------------------------------------------------------------------- GEMM ---
// C[M, NCOLS] = A[M, 768] * W[NCOLS, 768]^T + bias.  Block tile 64x256,
// 8 waves (2x4), each wave 32x64 (2x4 wmma).  W tile staged in LDS as
// [col][k] so B-fragments are contiguous ds_load_b128 pairs.
template <int MODE>
__global__ __launch_bounds__(256) void wmma_gemm(
    const unsigned short* __restrict__ A, const unsigned short* __restrict__ W,
    const float* __restrict__ bias, unsigned short* __restrict__ o_q,
    unsigned short* __restrict__ o_k, unsigned short* __restrict__ o_v,
    float* __restrict__ o_f) {
    __shared__ unsigned short bt[256 * 32];   // 16 KB weight tile

    const int tid  = threadIdx.x;
    const int w    = tid >> 5, lane = tid & 31;
    const int half = lane >> 4, ln = lane & 15;
    const int m0   = blockIdx.y * 64 + (w >> 2) * 32;
    const int nblk = blockIdx.x * 256;
    const int n0   = nblk + (w & 3) * 64;

    v8f acc[2][4] = {};

    for (int k0 = 0; k0 < C; k0 += 32) {
        {   // stage W tile: 256 cols x 32 k  (each thread moves one 64B column slice)
            const u16x8* src = (const u16x8*)(W + (size_t)(nblk + tid) * C + k0);
            u16x8* dst = (u16x8*)(bt + tid * 32);
            dst[0] = src[0];
            dst[1] = src[1];
            dst[2] = src[2];
            dst[3] = src[3];
        }
        __syncthreads();

        Frag af[2], bf[4];
#pragma unroll
        for (int mi = 0; mi < 2; mi++) {
            af[mi] = load_afrag(A, C, m0 + mi * 16 + ln, k0, half);
            __builtin_prefetch(A + (size_t)(m0 + mi * 16 + ln) * C + k0 + 128, 0, 0);
        }
#pragma unroll
        for (int ni = 0; ni < 4; ni++) {
            const unsigned short* p = bt + ((w & 3) * 64 + ni * 16 + ln) * 32 + half * 16;
            bf[ni].u[0] = *(const u16x8*)p;
            bf[ni].u[1] = *(const u16x8*)(p + 8);
        }
#pragma unroll
        for (int mi = 0; mi < 2; mi++)
#pragma unroll
            for (int ni = 0; ni < 4; ni++)
                acc[mi][ni] = wmma_bf16(af[mi], bf[ni], acc[mi][ni]);
        __syncthreads();
    }

#pragma unroll
    for (int mi = 0; mi < 2; mi++) {
#pragma unroll
        for (int ni = 0; ni < 4; ni++) {
            const int colg = n0 + ni * 16 + ln;
            const float bv = bias[colg];
#pragma unroll
            for (int r = 0; r < 8; r++) {
                const int row = m0 + mi * 16 + r + half * 8;
                const float val = acc[mi][ni][r] + bv;
                if (MODE == 0) {  // QKV epilogue: split & relayout
                    const int which = colg / C;
                    const int rem   = colg - which * C;
                    const int h     = rem >> 6, hd = rem & 63;
                    const int b     = row >> 10, n = row & (N - 1);
                    const size_t bh = (size_t)(b * H + h);
                    if (which == 0)      o_q[(bh * N + n) * HD + hd] = f2bf(val * 0.125f); // *HD^-0.5
                    else if (which == 1) o_k[(bh * N + n) * HD + hd] = f2bf(val);
                    else                 o_v[(bh * HD + hd) * N + n] = f2bf(val);  // V^T
                } else {          // proj epilogue: fp32 out
                    o_f[(size_t)row * C + colg] = val;
                }
            }
        }
    }
}

// -------------------------------------------------------- flash attention ---
// grid = B*H*(N/64); 4 waves/block; wave owns 16 query rows; online softmax
// over 32-key tiles.  Row sums come from an extra WMMA against a ones matrix
// (every column of P@Ones holds the row sum -> no shuffle, no broadcast);
// only the row-max still uses cross-lane reduction.  V fragments are issued
// at the top of each iteration so their latency hides behind the score WMMAs
// and the softmax VALU block.
__global__ __launch_bounds__(128) void attn_kernel(
    const unsigned short* __restrict__ Qb, const unsigned short* __restrict__ Kb,
    const unsigned short* __restrict__ Vt, const float* __restrict__ rpb,
    unsigned short* __restrict__ O1) {
    __shared__ unsigned short pl[4 * 16 * 32];  // per-wave P tile (q16 x k32)

    int t = blockIdx.x;
    const int qt = t & 15; t >>= 4;
    const int h  = t % H;
    const int b  = t / H;

    const int tid = threadIdx.x, w = tid >> 5, lane = tid & 31;
    const int half = lane >> 4, ln = lane & 15;
    const int q0 = qt * 64 + w * 16;

    const unsigned short* Q = Qb + (size_t)(b * H + h) * N * HD;
    const unsigned short* K = Kb + (size_t)(b * H + h) * N * HD;
    const unsigned short* V = Vt + (size_t)(b * H + h) * HD * N;
    const float* rp = rpb + (size_t)h * 63 * 63;

    Frag qf[2];
    qf[0] = load_afrag(Q, HD, q0 + ln, 0, half);
    qf[1] = load_afrag(Q, HD, q0 + ln, 32, half);

    Frag onesf;   // B-matrix of all 1.0bf16: P@Ones -> row sums in every lane
#pragma unroll
    for (int i = 0; i < 8; i++) { onesf.u[0][i] = 0x3F80; onesf.u[1][i] = 0x3F80; }

    float mrow[8];
#pragma unroll
    for (int r = 0; r < 8; r++) mrow[r] = -3.0e38f;
    v8f oacc[5] = {};   // [0..3] = O tiles (hd 0..63), [4] = row-sum accumulator

    unsigned short* plw = pl + w * 512;

    for (int kt = 0; kt < N; kt += 32) {
        // issue V fragment loads first: no intra-iteration dependency, their
        // latency hides behind the 4 score WMMAs + softmax VALU work below.
        Frag vf[4];
#pragma unroll
        for (int t4 = 0; t4 < 4; t4++) {
            const unsigned short* vp = V + (size_t)(t4 * 16 + ln) * N + kt + half * 16;
            vf[t4].u[0] = *(const u16x8*)vp;
            vf[t4].u[1] = *(const u16x8*)(vp + 8);
        }
        // scores: S[16 x 32] = Q[16 x 64] * K^T
        v8f s[2] = {};
#pragma unroll
        for (int ct = 0; ct < 2; ct++) {
            Frag kf0 = load_bfrag(K, HD, kt + ct * 16 + ln, 0, half);
            Frag kf1 = load_bfrag(K, HD, kt + ct * 16 + ln, 32, half);
            s[ct] = wmma_bf16(qf[0], kf0, s[ct]);
            s[ct] = wmma_bf16(qf[1], kf1, s[ct]);
        }
        // speculative prefetch of next key tile's K columns
        __builtin_prefetch(K + (size_t)(kt + 32 + ln) * HD, 0, 0);
        __builtin_prefetch(K + (size_t)(kt + 48 + ln) * HD, 0, 0);

        // bias add + online max (per-row stats live at VGPR index r)
        float p[2][8], alpha[8];
#pragma unroll
        for (int r = 0; r < 8; r++) {
            const int qg = q0 + r + half * 8;
            const int aq = qg >> 5, bq = qg & 31;
#pragma unroll
            for (int ct = 0; ct < 2; ct++) {
                const int kg = kt + ct * 16 + ln;
                s[ct][r] += rp[((kg >> 5) - aq + 31) * 63 + ((kg & 31) - bq + 31)];
            }
            float mx = fmaxf(s[0][r], s[1][r]);
#pragma unroll
            for (int off = 1; off < 16; off <<= 1) mx = fmaxf(mx, __shfl_xor(mx, off, 16));
            const float mn = fmaxf(mrow[r], mx);
            alpha[r] = __expf(mrow[r] - mn);
            mrow[r]  = mn;
            p[0][r]  = __expf(s[0][r] - mn);
            p[1][r]  = __expf(s[1][r] - mn);
        }
#pragma unroll
        for (int t5 = 0; t5 < 5; t5++)
#pragma unroll
            for (int r = 0; r < 8; r++) oacc[t5][r] *= alpha[r];

        // D-layout -> A-layout via wave-private LDS (DS ops are in-order per wave)
#pragma unroll
        for (int ct = 0; ct < 2; ct++)
#pragma unroll
            for (int r = 0; r < 8; r++)
                plw[(r + half * 8) * 32 + ct * 16 + ln] = f2bf(p[ct][r]);

        Frag pf;
        {
            const unsigned short* pp = plw + ln * 32 + half * 16;
            pf.u[0] = *(const u16x8*)pp;
            pf.u[1] = *(const u16x8*)(pp + 8);
        }
        // O += P[16 x 32] * V[32 x 64]; row-sum acc += P * Ones
#pragma unroll
        for (int t4 = 0; t4 < 4; t4++) oacc[t4] = wmma_bf16(pf, vf[t4], oacc[t4]);
        oacc[4] = wmma_bf16(pf, onesf, oacc[4]);
    }

    float inv[8];
#pragma unroll
    for (int r = 0; r < 8; r++) inv[r] = 1.0f / oacc[4][r];
#pragma unroll
    for (int t4 = 0; t4 < 4; t4++)
#pragma unroll
        for (int r = 0; r < 8; r++) {
            const int qg = q0 + r + half * 8;
            O1[((size_t)b * N + qg) * C + h * 64 + t4 * 16 + ln] = f2bf(oacc[t4][r] * inv[r]);
        }
}

// ---------------------------------------------------------------- launch ----
extern "C" void kernel_launch(void* const* d_in, const int* in_sizes, int n_in,
                              void* d_out, int out_size, void* d_ws, size_t ws_size,
                              hipStream_t stream) {
    (void)in_sizes; (void)n_in; (void)out_size; (void)ws_size;
    const float* x      = (const float*)d_in[0];
    const float* w_qkv  = (const float*)d_in[1];
    const float* b_qkv  = (const float*)d_in[2];
    const float* rpb    = (const float*)d_in[3];
    const float* w_proj = (const float*)d_in[4];
    const float* b_proj = (const float*)d_in[5];
    float* out = (float*)d_out;

    char* ws = (char*)d_ws;
    size_t off = 0;
    auto carve = [&](size_t bytes) {
        void* p = ws + off;
        off += (bytes + 255) & ~(size_t)255;
        return p;
    };
    unsigned short* xb  = (unsigned short*)carve((size_t)M * C * 2);
    unsigned short* wqb = (unsigned short*)carve((size_t)C3 * C * 2);
    unsigned short* wpb = (unsigned short*)carve((size_t)C * C * 2);
    unsigned short* Qb  = (unsigned short*)carve((size_t)Bsz * H * N * HD * 2);
    unsigned short* Kb  = (unsigned short*)carve((size_t)Bsz * H * N * HD * 2);
    unsigned short* Vt  = (unsigned short*)carve((size_t)Bsz * H * HD * N * 2);
    unsigned short* O1  = (unsigned short*)carve((size_t)M * C * 2);

    convert_bf16<<<2048, 256, 0, stream>>>(x, w_qkv, w_proj, xb, wqb, wpb);
    wmma_gemm<0><<<dim3(C3 / 256, M / 64), 256, 0, stream>>>(
        xb, wqb, b_qkv, Qb, Kb, Vt, nullptr);
    attn_kernel<<<Bsz * H * (N / 64), 128, 0, stream>>>(Qb, Kb, Vt, rpb, O1);
    wmma_gemm<1><<<dim3(C / 256, M / 64), 256, 0, stream>>>(
        O1, wpb, b_proj, nullptr, nullptr, nullptr, out);
}
